// RNN_7370163880375
// MI455X (gfx1250) — matrix operand order repaired and checked by reading, not compiled, not run on hardware
//
#include <hip/hip_runtime.h>
#include <hip/hip_bf16.h>

// RNN (tanh), B=256, T=4096, H=64, I=1, + Linear(64->1) per step.
// hᵀ-orientation recurrence on v_wmma_f32_16x16x32_bf16, f32 accumulate.
// Double-buffered h/partials -> ONE barrier per time step.
// Split WMMA accumulators -> no WMMA->WMMA dependency in the serial chain.

#define Hh   64
#define Tt   4096
#define Bb   256
#define CT   64            // time steps staged per LDS chunk
#define NT   (Tt / CT)
#define ROWH 72            // h_sh row stride in halfwords (64 + 8 pad; 144 B: 16B-aligned, conflict-free)
#define HBUF (16 * ROWH)   // one h buffer (halfwords)
#define XROW 68            // x_sh / ob_sh row stride in floats (272 B: 16B-aligned, conflict-free)

typedef __attribute__((ext_vector_type(16))) __bf16 v16bf;
typedef __attribute__((ext_vector_type(8)))  float  v8f;

union VB { unsigned u[8]; uint4 q[2]; v16bf v; };

// f32x2 -> packed bf16 (round-half-up: add 0x8000 to raw bits, keep high half).
// Single v_perm_b32 packs both high halves.
__device__ __forceinline__ unsigned pk_bf16(float a, float b) {
  unsigned ua = __builtin_bit_cast(unsigned, a) + 0x8000u;
  unsigned ub = __builtin_bit_cast(unsigned, b) + 0x8000u;
#if __has_builtin(__builtin_amdgcn_perm)
  return __builtin_amdgcn_perm(ub, ua, 0x07060302u);   // D = {ub[3],ub[2],ua[3],ua[2]}
#else
  return (ua >> 16) | (ub & 0xffff0000u);
#endif
}

__device__ __forceinline__ float fast_tanh(float x) {
#if __has_builtin(__builtin_amdgcn_tanhf)
  return __builtin_amdgcn_tanhf(x);                    // gfx1250 v_tanh_f32 (TRANS)
#else
  float e = __builtin_amdgcn_exp2f(x * 2.8853900817779268f);   // e^{2x}
  float r = __builtin_amdgcn_rcpf(1.0f + e);
  return 1.0f - 2.0f * r;
#endif
}

__global__ void __launch_bounds__(128)
rnn_wmma_kernel(const float* __restrict__ x,     const float* __restrict__ h0,
                const float* __restrict__ W_ih,  const float* __restrict__ W_hh,
                const float* __restrict__ b_ih,  const float* __restrict__ b_hh,
                const float* __restrict__ W_out, const float* __restrict__ b_out,
                float* __restrict__ outs,        float* __restrict__ h_last)
{
  __shared__ unsigned short h_sh[2 * HBUF];    // bf16 h[buf][batch16][64+pad]
  __shared__ float x_sh [16 * XROW];           // staged x chunk
  __shared__ float ob_sh[16 * XROW];           // staged outs chunk
  __shared__ float part_sh[2][16 * 8];         // out partials [buf][batch][slot]

  const int tid  = threadIdx.x;
  const int wave = tid >> 5, lane = tid & 31;
  const int l16  = lane & 15, hi = lane >> 4;          // batch col / half-wave
  const int hbase = 16 * wave + 8 * hi;                // this lane's hidden base (D rows)
  const int bbase = blockIdx.x * 16;                   // batch rows of this WG
  const bool red0 = (wave == 0) && (lane < 16);        // out-reduction lanes

  // ---- per-lane constants (one-time) ----
  float vw[8], vb[8], vo[8];
#pragma unroll
  for (int j = 0; j < 8; ++j) {
    vw[j] = W_ih[hbase + j];
    vb[j] = b_ih[hbase + j] + b_hh[hbase + j];
    vo[j] = W_out[hbase + j];
  }
  const float bo = b_out[0];

  // ---- A operands: W_hh rows [16*wave, 16*wave+16), bf16 A-layout, K chunks 0-31 / 32-63 ----
  v16bf A0, A1;
  {
    const float* wr = W_hh + (16 * wave + l16) * Hh;   // lane -> M row
    VB a0, a1;
#pragma unroll
    for (int k = 0; k < 4; ++k) {                      // elems 0-7: K = 8*hi + e ; elems 8-15: K = 16 + 8*hi + e
      a0.u[k]     = pk_bf16(wr[ 8*hi      + 2*k], wr[ 8*hi      + 2*k + 1]);
      a0.u[4 + k] = pk_bf16(wr[16 + 8*hi  + 2*k], wr[16 + 8*hi  + 2*k + 1]);
      a1.u[k]     = pk_bf16(wr[32 + 8*hi  + 2*k], wr[32 + 8*hi  + 2*k + 1]);
      a1.u[4 + k] = pk_bf16(wr[48 + 8*hi  + 2*k], wr[48 + 8*hi  + 2*k + 1]);
    }
    A0 = a0.v; A1 = a1.v;
  }

  // ---- init h_sh buf 0 from h_state (f32 -> bf16), cooperative ----
  {
    const int r = tid >> 3, c0 = (tid & 7) * 8;
    const float* hp = h0 + (bbase + r) * Hh + c0;
    uint4 q;
    q.x = pk_bf16(hp[0], hp[1]); q.y = pk_bf16(hp[2], hp[3]);
    q.z = pk_bf16(hp[4], hp[5]); q.w = pk_bf16(hp[6], hp[7]);
    *(uint4*)&h_sh[r * ROWH + c0] = q;
  }

  int p = 0;                                            // h read-buf / part write-buf
  float th[8];                                          // last tanh (kept for h_last)
  const v8f zc = {};                                    // zero C for the second WMMA

  for (int tc = 0; tc < NT; ++tc) {
    const int t0 = tc * CT;

    // deferred reduction for the last step of the previous chunk
    if (tc > 0 && red0) {
      const float4* pp = (const float4*)&part_sh[p ^ 1][l16 * 8];
      float4 a = pp[0], b = pp[1];
      a.x += b.x; a.y += b.y; a.z += b.z; a.w += b.w;
      ob_sh[l16 * XROW + (CT - 1)] = (a.x + a.y) + (a.z + a.w);
    }
    // stage x[bbase..+16)[t0..t0+CT) into LDS (coalesced)
    {
      const int r = tid >> 3, i0 = (tid & 7) * 8;
      const float4* xp = (const float4*)(x + (size_t)(bbase + r) * Tt + t0 + i0);
      *(float4*)&x_sh[r * XROW + i0]     = xp[0];
      *(float4*)&x_sh[r * XROW + i0 + 4] = xp[1];
    }
    __syncthreads();

    // flush previous outs chunk (coalesced)
    if (tc > 0) {
      const int r = tid >> 3, i0 = (tid & 7) * 8;
      float4 o0 = *(float4*)&ob_sh[r * XROW + i0];
      float4 o1 = *(float4*)&ob_sh[r * XROW + i0 + 4];
      o0.x += bo; o0.y += bo; o0.z += bo; o0.w += bo;
      o1.x += bo; o1.y += bo; o1.z += bo; o1.w += bo;
      float4* op = (float4*)(outs + (size_t)(bbase + r) * Tt + (t0 - CT) + i0);
      op[0] = o0; op[1] = o1;
    }

    for (int tt = 0; tt < CT; ++tt) {
      // deferred out-reduction for step t-1 (2x ds_load_b128 + packed adds)
      if (tt > 0 && red0) {
        const float4* pp = (const float4*)&part_sh[p ^ 1][l16 * 8];
        float4 a = pp[0], b = pp[1];
        a.x += b.x; a.y += b.y; a.z += b.z; a.w += b.w;
        ob_sh[l16 * XROW + (tt - 1)] = (a.x + a.y) + (a.z + a.w);
      }

      // C init: x_t * W_ih + (b_ih + b_hh)   (exact f32)
      const float xv = x_sh[l16 * XROW + tt];
      v8f c0;
#pragma unroll
      for (int j = 0; j < 8; ++j) c0[j] = fmaf(vw[j], xv, vb[j]);

      // B operands (h_prevᵀ) from h_sh[p]: lane = batch col, K along regs
      VB b0, b1;
      const uint4* hrow = (const uint4*)&h_sh[p * HBUF + l16 * ROWH];
      b0.q[0] = hrow[hi];     b0.q[1] = hrow[2 + hi];      // K 0-31
      b1.q[0] = hrow[4 + hi]; b1.q[1] = hrow[6 + hi];  // K 32-63

      // independent WMMAs (no D->C chain), merged with packed adds
      c0 = __builtin_amdgcn_wmma_f32_16x16x32_bf16(false, A0, false, b0.v, (short)0, c0, false, false);
      v8f c1 = __builtin_amdgcn_wmma_f32_16x16x32_bf16(false, A1, false, b1.v, (short)0, zc, false, false);
      v8f c = c0 + c1;

      // tanh + output-projection partial (pairwise tree, f32)
#pragma unroll
      for (int j = 0; j < 8; ++j) th[j] = fast_tanh(c[j]);
      float pa = fmaf(th[1], vo[1], th[0] * vo[0]);
      float pb = fmaf(th[3], vo[3], th[2] * vo[2]);
      float pc = fmaf(th[5], vo[5], th[4] * vo[4]);
      float pd = fmaf(th[7], vo[7], th[6] * vo[6]);
      float po = (pa + pb) + (pc + pd);

      // publish new h (bf16, one b128 per lane) into h_sh[p^1] + out partial
      uint4 q;
      q.x = pk_bf16(th[0], th[1]); q.y = pk_bf16(th[2], th[3]);
      q.z = pk_bf16(th[4], th[5]); q.w = pk_bf16(th[6], th[7]);
      *(uint4*)&h_sh[(p ^ 1) * HBUF + l16 * ROWH + hbase] = q;
      part_sh[p][l16 * 8 + 2 * wave + hi] = po;

      __syncthreads();                                  // single barrier per step
      p ^= 1;
    }
  }

  // ---- epilogue: reduce final step's partials, flush last outs chunk ----
  if (red0) {
    const float4* pp = (const float4*)&part_sh[p ^ 1][l16 * 8];
    float4 a = pp[0], b = pp[1];
    a.x += b.x; a.y += b.y; a.z += b.z; a.w += b.w;
    ob_sh[l16 * XROW + (CT - 1)] = (a.x + a.y) + (a.z + a.w);
  }
  __syncthreads();
  {
    const int r = tid >> 3, i0 = (tid & 7) * 8;
    float4 o0 = *(float4*)&ob_sh[r * XROW + i0];
    float4 o1 = *(float4*)&ob_sh[r * XROW + i0 + 4];
    o0.x += bo; o0.y += bo; o0.z += bo; o0.w += bo;
    o1.x += bo; o1.y += bo; o1.z += bo; o1.w += bo;
    float4* op = (float4*)(outs + (size_t)(bbase + r) * Tt + (Tt - CT) + i0);
    op[0] = o0; op[1] = o1;
  }

  // ---- h_last (f32, from last step's registers) ----
  {
    float* hp = h_last + (bbase + l16) * Hh + hbase;
    float4 q0 = { th[0], th[1], th[2], th[3] };
    float4 q1 = { th[4], th[5], th[6], th[7] };
    ((float4*)hp)[0] = q0; ((float4*)hp)[1] = q1;
  }
}

extern "C" void kernel_launch(void* const* d_in, const int* in_sizes, int n_in,
                              void* d_out, int out_size, void* d_ws, size_t ws_size,
                              hipStream_t stream) {
  (void)in_sizes; (void)n_in; (void)out_size; (void)d_ws; (void)ws_size;
  const float* x     = (const float*)d_in[0];
  const float* h0    = (const float*)d_in[1];
  const float* W_ih  = (const float*)d_in[2];
  const float* W_hh  = (const float*)d_in[3];
  const float* b_ih  = (const float*)d_in[4];
  const float* b_hh  = (const float*)d_in[5];
  const float* W_out = (const float*)d_in[6];
  const float* b_out = (const float*)d_in[7];
  float* outs   = (float*)d_out;                 // [B,T,1]
  float* h_last = outs + (size_t)Bb * Tt;        // [1,B,H]
  rnn_wmma_kernel<<<Bb / 16, 128, 0, stream>>>(x, h0, W_ih, W_hh, b_ih, b_hh,
                                               W_out, b_out, outs, h_last);
}
